// HeteroGNN_4707284157148
// MI455X (gfx1250) — matrix operand order repaired and checked by reading, not compiled, not run on hardware
//
#include <hip/hip_runtime.h>
#include <math.h>

// ---------------- types for WMMA ----------------
typedef __attribute__((ext_vector_type(16))) __bf16 v16bf;
typedef __attribute__((ext_vector_type(8)))  float  v8f;

// fp32 -> bf16 round-to-nearest-even (manual, avoids relying on native cvt)
static __device__ __forceinline__ __bf16 f2bf(float f) {
    unsigned u = __builtin_bit_cast(unsigned, f);
    unsigned r = u + 0x7FFFu + ((u >> 16) & 1u);
    unsigned short s = (unsigned short)(r >> 16);
    return __builtin_bit_cast(__bf16, s);
}
static __device__ __forceinline__ float bf2f(__bf16 h) {
    unsigned u = ((unsigned)__builtin_bit_cast(unsigned short, h)) << 16;
    return __builtin_bit_cast(float, u);
}

// ---------------- small helper kernels ----------------

// T[c*R + r] = split(W[r*C + c])  (transpose + bf16 hi/lo split of a weight matrix)
__global__ void k_transpose_split(const float* __restrict__ W, int R, int C,
                                  __bf16* __restrict__ Thi, __bf16* __restrict__ Tlo) {
    int idx = blockIdx.x * blockDim.x + threadIdx.x;
    if (idx >= R * C) return;
    int r = idx / C, c = idx % C;
    float v = W[idx];
    __bf16 h = f2bf(v);
    Thi[(size_t)c * R + r] = h;
    Tlo[(size_t)c * R + r] = f2bf(v - bf2f(h));
}

// v[d] = sum_h W[d*128+h] * a[h]   (128 threads)
__global__ void k_matvec128(const float* __restrict__ W, const float* __restrict__ a,
                            float* __restrict__ v) {
    int d = threadIdx.x;
    float s = 0.f;
    #pragma unroll 8
    for (int h = 0; h < 128; ++h) s += W[d * 128 + h] * a[h];
    v[d] = s;
}

// s[j] = x[j,:128] . v   (one wave per row)
__global__ __launch_bounds__(256)
void k_rowdot(const float* __restrict__ X, int n, const float* __restrict__ v,
              float* __restrict__ s) {
    int wid  = (blockIdx.x * blockDim.x + threadIdx.x) >> 5;
    int lane = threadIdx.x & 31;
    if (wid >= n) return;
    const float4* xp = (const float4*)(X + (size_t)wid * 128);
    const float4* vp = (const float4*)v;
    float4 x = xp[lane], w = vp[lane];
    float d = x.x * w.x + x.y * w.y + x.z * w.z + x.w * w.w;
    for (int o = 16; o > 0; o >>= 1) d += __shfl_xor(d, o, 32);
    if (lane == 0) s[wid] = d;
}

__global__ void k_init(float* __restrict__ agg, long long agg_n,
                       float* __restrict__ m, float* __restrict__ denom, int n_dst) {
    long long i = (long long)blockIdx.x * blockDim.x + threadIdx.x;
    if (i < agg_n) agg[i] = 0.f;
    if (i < n_dst) { m[i] = -__builtin_inff(); denom[i] = 0.f; }
}

// e = leaky_relu(s_src[src] + s_dst[dst]); atomic segment max into m[dst]
__global__ void k_edge_logit(const int* __restrict__ src, const int* __restrict__ dst, int E,
                             const float* __restrict__ s_src, const float* __restrict__ s_dst,
                             float* __restrict__ e_ws, float* __restrict__ m) {
    int i = blockIdx.x * blockDim.x + threadIdx.x;
    if (i >= E) return;
    float e = s_src[src[i]] + s_dst[dst[i]];
    e = e > 0.f ? e : 0.2f * e;
    e_ws[i] = e;
    float* p = m + dst[i];
    if (e >= 0.f) atomicMax((int*)p,          __float_as_int(e));
    else          atomicMin((unsigned int*)p, __float_as_uint(e));
}

__global__ void k_fix_m(float* __restrict__ m, int n) {
    int i = blockIdx.x * blockDim.x + threadIdx.x;
    if (i >= n) return;
    float v = m[i];
    if (!(v > -3.0e38f)) m[i] = 0.f;   // empty segment: max stayed at -inf
}

__global__ void k_edge_exp(const int* __restrict__ dst, int E, const float* __restrict__ m,
                           float* __restrict__ e_ws, float* __restrict__ denom) {
    int i = blockIdx.x * blockDim.x + threadIdx.x;
    if (i >= E) return;
    int d = dst[i];
    float ex = __expf(e_ws[i] - m[d]);
    e_ws[i] = ex;
    atomicAdd(denom + d, ex);
}

// one wave per edge: agg[dst,:] += alpha * h_src[src,:]
__global__ __launch_bounds__(256)
void k_edge_scatter(const int* __restrict__ src, const int* __restrict__ dst, int E,
                    const float* __restrict__ e_ws, const float* __restrict__ denom,
                    const float* __restrict__ h_src, float* __restrict__ agg) {
    int wid  = (blockIdx.x * blockDim.x + threadIdx.x) >> 5;
    int lane = threadIdx.x & 31;
    if (wid >= E) return;
    int s = src[wid], d = dst[wid];
    float w = e_ws[wid] / denom[d];
    const float4* hp = (const float4*)(h_src + (size_t)s * 128);
    float4 v = hp[lane];
    float* op = agg + (size_t)d * 128 + lane * 4;
    atomicAdd(op + 0, w * v.x);
    atomicAdd(op + 1, w * v.y);
    atomicAdd(op + 2, w * v.z);
    atomicAdd(op + 3, w * v.w);
}

// ---------------- WMMA GEMM, bf16x3 fp32-emulation ----------------
// C[M,N] = act(A[M,K] (+biasK)) @ B[K,N] (+biasN), B given pre-transposed [N,K]
// as bf16 hi/lo. A is fp32; split to bf16 hi/lo in-register per fragment.
// One wave computes 16 rows x N cols. Wave32; K multiple of 32; M multiple of 16.
template<int NTILES, bool FUSE_BIAS_RELU, bool OUT_BIAS>
__global__ __launch_bounds__(256)
void gemm_bf16x3(const float* __restrict__ A, int M, int K,
                 const __bf16* __restrict__ BThi, const __bf16* __restrict__ BTlo,
                 const float* __restrict__ biasK, const float* __restrict__ biasN,
                 float* __restrict__ C) {
    const int N   = NTILES * 16;
    int wave  = (blockIdx.x * blockDim.x + threadIdx.x) >> 5;
    int lane  = threadIdx.x & 31;
    int mbase = wave * 16;
    if (mbase >= M) return;

    v8f acc[NTILES];
    #pragma unroll
    for (int t = 0; t < NTILES; ++t)
        #pragma unroll
        for (int j = 0; j < 8; ++j) acc[t][j] = 0.f;

    const int rowA  = mbase + (lane & 15);
    const int koffA = (lane >> 4) << 3;   // A frag: K chunks {0-7,16-23} / {8-15,24-31}
    const int koffB = (lane >> 4) << 3;   // B frag: K chunks {0-7,16-23} / {8-15,24-31}

    for (int k0 = 0; k0 < K; k0 += 32) {
        // ---- A fragment: load fp32, optional bias+relu, split hi/lo ----
        const float4* pa = (const float4*)(A + (size_t)rowA * K + k0 + koffA);
        float4 a0 = pa[0], a1 = pa[1], a2 = pa[4], a3 = pa[5];
        float ta[16] = { a0.x,a0.y,a0.z,a0.w, a1.x,a1.y,a1.z,a1.w,
                         a2.x,a2.y,a2.z,a2.w, a3.x,a3.y,a3.z,a3.w };
        if (FUSE_BIAS_RELU) {
            const float4* pb = (const float4*)(biasK + k0 + koffA);
            float4 b0 = pb[0], b1 = pb[1], b2 = pb[4], b3 = pb[5];
            float tb[16] = { b0.x,b0.y,b0.z,b0.w, b1.x,b1.y,b1.z,b1.w,
                             b2.x,b2.y,b2.z,b2.w, b3.x,b3.y,b3.z,b3.w };
            #pragma unroll
            for (int i = 0; i < 16; ++i) {
                float t = ta[i] + tb[i];
                ta[i] = t > 0.f ? t : 0.f;
            }
        }
        v16bf ahi, alo;
        #pragma unroll
        for (int i = 0; i < 16; ++i) {
            __bf16 h = f2bf(ta[i]);
            ahi[i] = h;
            alo[i] = f2bf(ta[i] - bf2f(h));
        }
        // ---- per N-tile: load B frags, 3 WMMAs (hi*hi + hi*lo + lo*hi) ----
        #pragma unroll
        for (int t = 0; t < NTILES; ++t) {
            int n = t * 16 + (lane & 15);
            typedef __attribute__((ext_vector_type(8))) __bf16 v8bf_;
            const v8bf_* pbh = (const v8bf_*)(BThi + (size_t)n * K + k0 + koffB);
            const v8bf_* pbl = (const v8bf_*)(BTlo + (size_t)n * K + k0 + koffB);
            v16bf bhi = __builtin_shufflevector(pbh[0], pbh[2], 0,1,2,3,4,5,6,7,8,9,10,11,12,13,14,15);
            v16bf blo = __builtin_shufflevector(pbl[0], pbl[2], 0,1,2,3,4,5,6,7,8,9,10,11,12,13,14,15);
            acc[t] = __builtin_amdgcn_wmma_f32_16x16x32_bf16(
                         false, ahi, false, bhi, (short)0, acc[t], false, false);
            acc[t] = __builtin_amdgcn_wmma_f32_16x16x32_bf16(
                         false, ahi, false, blo, (short)0, acc[t], false, false);
            acc[t] = __builtin_amdgcn_wmma_f32_16x16x32_bf16(
                         false, alo, false, bhi, (short)0, acc[t], false, false);
        }
    }
    // ---- store (C/D layout: lane 0-15 -> M=mbase+r, lane 16-31 -> M=mbase+8+r)
    int ncol0 = lane & 15;
    int rbase = mbase + ((lane >> 4) << 3);
    #pragma unroll
    for (int t = 0; t < NTILES; ++t) {
        int n = t * 16 + ncol0;
        float bn = OUT_BIAS ? biasN[n] : 0.f;
        #pragma unroll
        for (int r = 0; r < 8; ++r)
            C[(size_t)(rbase + r) * N + n] = acc[t][r] + bn;
    }
}

// ---------------- launch ----------------
extern "C" void kernel_launch(void* const* d_in, const int* in_sizes, int n_in,
                              void* d_out, int out_size, void* d_ws, size_t ws_size,
                              hipStream_t stream) {
    const float* x_label  = (const float*)d_in[0];
    const float* x_attr   = (const float*)d_in[1];
    const float* W_src_al = (const float*)d_in[7];
    const float* W_dst_al = (const float*)d_in[8];
    const float* a_src_al = (const float*)d_in[9];
    const float* a_dst_al = (const float*)d_in[10];
    const float* b_al     = (const float*)d_in[11];
    const float* W_lin    = (const float*)d_in[12];
    const float* b_lin    = (const float*)d_in[13];
    const int*   e_src    = (const int*)d_in[16];   // edge_src_al (attr side)
    const int*   e_dst    = (const int*)d_in[17];   // edge_dst_al (label side)

    const int N_L = in_sizes[0] / 128;
    const int N_A = in_sizes[1] / 128;
    const int E   = in_sizes[16];

    // workspace carve-out (256B aligned slices)
    char* ws = (char*)d_ws;
    size_t off = 0;
    auto carve = [&](size_t bytes) -> void* {
        void* p = ws + off;
        off = (off + bytes + 255) & ~(size_t)255;
        return p;
    };
    __bf16* WsT_hi = (__bf16*)carve(128 * 128 * sizeof(__bf16));
    __bf16* WsT_lo = (__bf16*)carve(128 * 128 * sizeof(__bf16));
    __bf16* WlT_hi = (__bf16*)carve(64 * 128 * sizeof(__bf16));
    __bf16* WlT_lo = (__bf16*)carve(64 * 128 * sizeof(__bf16));
    float*  v_src  = (float*)carve(128 * sizeof(float));
    float*  v_dst  = (float*)carve(128 * sizeof(float));
    float*  s_src  = (float*)carve((size_t)N_A * sizeof(float));
    float*  s_dst  = (float*)carve((size_t)N_L * sizeof(float));
    float*  h_src  = (float*)carve((size_t)N_A * 128 * sizeof(float));
    float*  e_ws   = (float*)carve((size_t)E * sizeof(float));
    float*  m_ws   = (float*)carve((size_t)N_L * sizeof(float));
    float*  den_ws = (float*)carve((size_t)N_L * sizeof(float));
    float*  agg    = (float*)carve((size_t)N_L * 128 * sizeof(float));
    (void)ws_size; (void)n_in; (void)out_size;

    // 1) weight prep: transpose + bf16 split; attention vectors
    k_transpose_split<<<(128 * 128 + 255) / 256, 256, 0, stream>>>(W_src_al, 128, 128, WsT_hi, WsT_lo);
    k_transpose_split<<<(128 * 64 + 255) / 256, 256, 0, stream>>>(W_lin, 128, 64, WlT_hi, WlT_lo);
    k_matvec128<<<1, 128, 0, stream>>>(W_src_al, a_src_al, v_src);
    k_matvec128<<<1, 128, 0, stream>>>(W_dst_al, a_dst_al, v_dst);

    // 2) per-node attention scalars (mat-vec instead of full h_dst GEMM)
    k_rowdot<<<(N_A * 32 + 255) / 256, 256, 0, stream>>>(x_attr, N_A, v_src, s_src);
    k_rowdot<<<(N_L * 32 + 255) / 256, 256, 0, stream>>>(x_label, N_L, v_dst, s_dst);

    // 3) h_src = x_attr @ W_src_al  (WMMA, bf16x3)
    {
        int waves = N_A / 16;
        gemm_bf16x3<8, false, false><<<(waves + 7) / 8, 256, 0, stream>>>(
            x_attr, N_A, 128, WsT_hi, WsT_lo, nullptr, nullptr, h_src);
    }

    // 4) init segment-softmax state + aggregation buffer
    {
        long long agg_n = (long long)N_L * 128;
        k_init<<<(unsigned)((agg_n + 255) / 256), 256, 0, stream>>>(agg, agg_n, m_ws, den_ws, N_L);
    }

    // 5) segment softmax + scatter
    k_edge_logit<<<(E + 255) / 256, 256, 0, stream>>>(e_src, e_dst, E, s_src, s_dst, e_ws, m_ws);
    k_fix_m<<<(N_L + 255) / 256, 256, 0, stream>>>(m_ws, N_L);
    k_edge_exp<<<(E + 255) / 256, 256, 0, stream>>>(e_dst, E, m_ws, e_ws, den_ws);
    k_edge_scatter<<<(E * 32 + 255) / 256, 256, 0, stream>>>(e_src, e_dst, E, e_ws, den_ws, h_src, agg);

    // 6) out = relu(agg + b_al) @ W_lin + b_lin  (WMMA, fused bias/relu epilogue)
    {
        int waves = N_L / 16;
        gemm_bf16x3<4, true, true><<<(waves + 7) / 8, 256, 0, stream>>>(
            agg, N_L, 128, WlT_hi, WlT_lo, b_al, b_lin, (float*)d_out);
    }
}
